// DualVSSBlock_70514773066479
// MI455X (gfx1250) — compile-verified, matrix operand order checked by reading, not run on hardware
//
#include <hip/hip_runtime.h>
#include <hip/hip_bf16.h>

// ---------------------------------------------------------------------------
// DualVSSBlock for MI455X (gfx1250, wave32, WMMA + Tensor Data Mover).
// B=4 H=W=48 C=192, per-branch dm=96 d=192 n=16 r=6, L=2304.
// Dense projections -> v_wmma_f32_16x16x32_bf16 (f32 accumulate).
// GEMM tiles staged to LDS via TDM (tensor_load_to_lds) when the 5-arg
// builtin toolchain (ROCm 7.2) is detected; manual global->LDS otherwise.
// Selective scan -> 1 wave per 2 channels (16 states/lane-group), shfl reduce.
// ---------------------------------------------------------------------------

typedef __bf16 bf16;
typedef __attribute__((ext_vector_type(16))) __bf16 v16bf;
typedef __attribute__((ext_vector_type(8)))  float  v8f;

#if __has_include(<hip/amd_detail/amd_gfx1250_TDM.h>)
// amdgpu-toolchain (clang-23): 6-arg tensor builtin -- use manual LDS staging.
#define VSS_TDM_5ARG 0
#else
// ROCm 7.2 (clang-22): 5-arg tensor builtin, probe-verified to compile.
#define VSS_TDM_5ARG 1
#endif

namespace cfg {
constexpr int B   = 4;
constexpr int H   = 48;
constexpr int W   = 48;
constexpr int C   = 192;   // full channels
constexpr int DM  = 96;    // per-branch model dim
constexpr int D   = 192;   // inner dim
constexpr int N   = 16;    // state dim
constexpr int R   = 6;     // dt rank
constexpr int L   = H * W;           // 2304
constexpr int RN2 = R + 2 * N;       // 38
constexpr int M   = B * L;           // 9216 pixels
}

// Direction index mapping: position in xc (row-major h*W+w) that feeds
// scan step l of direction k.  k0: row-major; k1: column-major; k2/k3 reversed.
__device__ __forceinline__ int mapPos(int k, int l) {
  int ll = (k >= 2) ? (cfg::L - 1 - l) : l;
  if (k & 1) { int h = ll % cfg::H; int w = ll / cfg::H; return h * cfg::W + w; }
  return ll;
}

#if VSS_TDM_5ARG
typedef __attribute__((ext_vector_type(4))) unsigned int u32x4_t;
typedef __attribute__((ext_vector_type(8))) int          i32x8_t;
typedef __attribute__((ext_vector_type(4))) int          i32x4_t;

// Issue one TDM 2D tile load: tile (tile_rows x 32 bf16) from a row-major
// (tensor_rows x tdim0) bf16 tensor with row stride `strideElems`, into LDS.
// D# fields per CDNA5 ISA Sec. 8.3/8.4 (2D tensor, groups
// 2/3 zero).  OOB rows (>= tensor_rows) read as zero -> free N padding.
__device__ __forceinline__ void tdm_load_2d(const bf16* gaddr, unsigned ldsOff,
                                            unsigned tdim0, unsigned tensorRows,
                                            unsigned strideElems,
                                            unsigned tileRows) {
  unsigned long long ga = (unsigned long long)(uintptr_t)gaddr;
  u32x4_t g0;
  g0[0] = 1u;                                      // count=1, user, no gather
  g0[1] = ldsOff;                                  // lds_addr (bytes)
  g0[2] = (unsigned)(ga & 0xFFFFFFFFu);            // global_addr[31:0]
  g0[3] = (unsigned)((ga >> 32) & 0x01FFFFFFu)     // global_addr[56:32]
        | (2u << 30);                              // type = 2 ("image")
  i32x8_t g1;
  g1[0] = (int)(1u << 16);                         // data_size=1 (2 bytes)
  g1[1] = (int)((tdim0 & 0xFFFFu) << 16);          // tensor_dim0[15:0]
  g1[2] = (int)(((tdim0 >> 16) & 0xFFFFu) |        // tensor_dim0[31:16]
                ((tensorRows & 0xFFFFu) << 16));   // tensor_dim1[15:0]
  g1[3] = (int)(((tensorRows >> 16) & 0xFFFFu) |   // tensor_dim1[31:16]
                (32u << 16));                      // tile_dim0 = 32 elems
  g1[4] = (int)(tileRows & 0xFFFFu);               // tile_dim1 (tile_dim2=0)
  g1[5] = (int)strideElems;                        // tensor_dim0_stride[31:0]
  g1[6] = 0;                                       // stride hi / dim1_stride lo
  g1[7] = 0;
  i32x4_t gz = {0, 0, 0, 0};
  __builtin_amdgcn_tensor_load_to_lds(g0, g1, gz, gz, 0);
}
#endif  // VSS_TDM_5ARG

// ---------------------------------------------------------------------------
// Weight conversion f32 -> bf16 (in_w, xproj_w, out_w)
// ---------------------------------------------------------------------------
__global__ void cvt_bf16_kernel(const float* __restrict__ a0, int n0,
                                const float* __restrict__ a1, int n1,
                                const float* __restrict__ a2, int n2,
                                bf16* __restrict__ o0, bf16* __restrict__ o1,
                                bf16* __restrict__ o2) {
  int i = blockIdx.x * 256 + threadIdx.x;
  if (i < n0) o0[i] = (bf16)a0[i];
  if (i < n1) o1[i] = (bf16)a1[i];
  if (i < n2) o2[i] = (bf16)a2[i];
}

// ---------------------------------------------------------------------------
// LayerNorm over 96 channels of one half of x, emit bf16 rows for WMMA GEMM.
// One wave per pixel (3 channels per lane), 8 pixels per block.
// ---------------------------------------------------------------------------
__global__ void ln_in_kernel(const float* __restrict__ x,
                             const float* __restrict__ w,
                             const float* __restrict__ bia,
                             bf16* __restrict__ xln, int brOff) {
  int wave = threadIdx.x >> 5, lane = threadIdx.x & 31;
  int p = blockIdx.x * 8 + wave;
  if (p >= cfg::M) return;
  const float* row = x + (size_t)p * cfg::C + brOff;
  float v0 = row[lane], v1 = row[lane + 32], v2 = row[lane + 64];
  float s = v0 + v1 + v2, sq = v0 * v0 + v1 * v1 + v2 * v2;
  for (int off = 16; off; off >>= 1) {
    s  += __shfl_xor(s,  off, 32);
    sq += __shfl_xor(sq, off, 32);
  }
  float m   = s / 96.f;
  float inv = rsqrtf(sq / 96.f - m * m + 1e-5f);
  bf16* o = xln + (size_t)p * cfg::DM;
  o[lane]      = (bf16)((v0 - m) * inv * w[lane]      + bia[lane]);
  o[lane + 32] = (bf16)((v1 - m) * inv * w[lane + 32] + bia[lane + 32]);
  o[lane + 64] = (bf16)((v2 - m) * inv * w[lane + 64] + bia[lane + 64]);
}

// ---------------------------------------------------------------------------
// Generic bf16 WMMA GEMM: out[m,n] = sum_k A[m,k] * Wt[n,k]  (Wt row-major N x K)
// Block = 256 threads (8 waves). Tile: 128(M) x 64(N), K-step 32.
// MODE 0: in-proj  (N=384) -> split cols into xc (f32) and z (f32)   [TDM tiles]
// MODE 1: x-proj   (N=38, grid.z = direction k, A rows remapped) -> xd[b,k,l,38]
// MODE 2: out-proj (N=96) -> residual add into d_out at channel offset [TDM tiles]
// ---------------------------------------------------------------------------
template <int MODE>
__global__ void gemm_wmma_kernel(const bf16* __restrict__ A, int K, int Ntot,
                                 const bf16* __restrict__ Wt,
                                 float* __restrict__ out0, float* __restrict__ out1,
                                 const float* __restrict__ resid,
                                 float* __restrict__ outf, int brOff) {
  __shared__ __align__(16) bf16 As[128 * 32];
  __shared__ __align__(16) bf16 Bs[64 * 32];

  const int tid  = threadIdx.x;
  const int wave = tid >> 5;
  const int lane = tid & 31;
  const int rowBlk = blockIdx.x * 128;
  const int colBlk = blockIdx.y * 64;
  const int kdir   = (MODE == 1) ? (int)blockIdx.z : 0;
  const bf16* Wm   = Wt + (MODE == 1 ? (size_t)kdir * cfg::RN2 * cfg::D : 0);

#if VSS_TDM_5ARG
  constexpr bool kUseTdm = (MODE != 1);
#else
  constexpr bool kUseTdm = false;
#endif

  v8f acc[4] = {};

  const int rsel  = lane & 15;
  const int khalf = (lane >> 4) & 1;

  for (int k0 = 0; k0 < K; k0 += 32) {
    if constexpr (kUseTdm) {
#if VSS_TDM_5ARG
      // One wave drives the Tensor Data Mover for both tiles, waits on
      // TENSORcnt, then the workgroup barrier releases the consumers.
      if (wave == 0) {
        tdm_load_2d(A + (size_t)rowBlk * K + k0,
                    (unsigned)(uintptr_t)As,
                    (unsigned)(K - k0), 128u, (unsigned)K, 128u);
        unsigned remN = (unsigned)(Ntot - colBlk);
        if (remN > 64u) remN = 64u;
        tdm_load_2d(Wm + (size_t)colBlk * K + k0,
                    (unsigned)(uintptr_t)Bs,
                    (unsigned)(K - k0), remN, (unsigned)K, 64u);
        __builtin_amdgcn_s_wait_tensorcnt(0);
      }
#endif
    } else {
      // --- manual stage A tile (128 x 32 bf16) into LDS ---
      {
        int r = tid >> 1, halfs = tid & 1;
        int gm = rowBlk + r;
        int srow;
        if (MODE == 1) {
          int b = gm / cfg::L, l = gm % cfg::L;
          srow = b * cfg::L + mapPos(kdir, l);
        } else {
          srow = gm;
        }
        const bf16* src = A + (size_t)srow * K + k0 + halfs * 16;
        if (k0 + 32 < K) __builtin_prefetch(src + 32, 0, 1);  // global_prefetch
        uint4 q0 = *(const uint4*)(src);
        uint4 q1 = *(const uint4*)(src + 8);
        *(uint4*)&As[r * 32 + halfs * 16]     = q0;
        *(uint4*)&As[r * 32 + halfs * 16 + 8] = q1;
      }
      // --- manual stage B tile (64 x 32 bf16, zero-padded past Ntot) ---
      {
        int nrow = tid >> 2, seg = tid & 3;
        int gn = colBlk + nrow;
        uint4 q = make_uint4(0u, 0u, 0u, 0u);
        if (gn < Ntot) q = *(const uint4*)(Wm + (size_t)gn * K + k0 + seg * 8);
        *(uint4*)&Bs[nrow * 32 + seg * 8] = q;
      }
    }
    __syncthreads();

    // --- A fragment: two contiguous 16B LDS runs per lane (ds_load_b128) ---
    const uint4* AsQ = (const uint4*)As;   // 8 bf16 per uint4; row = 4 uint4
    const uint4* BsQ = (const uint4*)Bs;
    union { uint4 q[2]; v16bf v; } af;
    {
      int row = wave * 16 + rsel;
      af.q[0] = AsQ[row * 4 + khalf];       // K = khalf*8 .. +7
      af.q[1] = AsQ[row * 4 + 2 + khalf];   // K = 16+khalf*8 .. +7
    }
    // --- 4 column tiles: B fragment + WMMA ---
#pragma unroll
    for (int ct = 0; ct < 4; ++ct) {
      union { uint4 q[2]; v16bf v; } bfg;
      int row = ct * 16 + rsel;
      bfg.q[0] = BsQ[row * 4 + khalf];
      bfg.q[1] = BsQ[row * 4 + 2 + khalf];
      acc[ct] = __builtin_amdgcn_wmma_f32_16x16x32_bf16(
          false, af.v, false, bfg.v, (short)0, acc[ct], false, false);
    }
    __syncthreads();
  }

  // --- epilogue: C/D layout -> VGPR j holds row (khalf?8:0)+j, col lane&15 ---
  const int rowOff = khalf ? 8 : 0;
  const int col    = lane & 15;
#pragma unroll
  for (int ct = 0; ct < 4; ++ct) {
    int gcol = colBlk + ct * 16 + col;
#pragma unroll
    for (int j = 0; j < 8; ++j) {
      int   gm = rowBlk + wave * 16 + rowOff + j;
      float v  = acc[ct][j];
      if (MODE == 0) {
        if (gcol < cfg::D) out0[(size_t)gm * cfg::D + gcol] = v;
        else               out1[(size_t)gm * cfg::D + (gcol - cfg::D)] = v;
      } else if (MODE == 1) {
        if (gcol < cfg::RN2) {
          int b = gm / cfg::L, l = gm % cfg::L;
          out0[((size_t)(b * 4 + kdir) * cfg::L + l) * cfg::RN2 + gcol] = v;
        }
      } else {
        if (gcol < Ntot) {
          size_t idx = (size_t)gm * cfg::C + brOff + gcol;
          outf[idx] = resid[idx] + v;
        }
      }
    }
  }
}

// ---------------------------------------------------------------------------
// Depthwise 3x3 conv (SAME) + bias + SiLU.  Dual-layout output:
//   xc_rm (B,L,D) bf16 for the x-proj GEMM, xc_cm (B,D,L) f32 for the scan.
// ---------------------------------------------------------------------------
__global__ void conv_silu_kernel(const float* __restrict__ xzc,
                                 const float* __restrict__ cw,
                                 const float* __restrict__ cb,
                                 bf16* __restrict__ xc_rm,
                                 float* __restrict__ xc_cm) {
  int id = blockIdx.x * 256 + threadIdx.x;
  int c = id % cfg::D;
  int p = id / cfg::D;
  if (p >= cfg::M) return;
  int b = p / cfg::L, l = p % cfg::L;
  int h = l / cfg::W, w = l % cfg::W;
  float acc = cb[c];
#pragma unroll
  for (int kh = 0; kh < 3; ++kh) {
    int hh = h + kh - 1;
    if (hh < 0 || hh >= cfg::H) continue;
#pragma unroll
    for (int kw = 0; kw < 3; ++kw) {
      int ww = w + kw - 1;
      if (ww < 0 || ww >= cfg::W) continue;
      acc += xzc[((size_t)(b * cfg::L + hh * cfg::W + ww)) * cfg::D + c] *
             cw[c * 9 + kh * 3 + kw];
    }
  }
  float s = acc / (1.f + __expf(-acc));  // SiLU
  xc_rm[(size_t)p * cfg::D + c]                   = (bf16)s;
  xc_cm[((size_t)(b * cfg::D + c)) * cfg::L + l]  = s;
}

// ---------------------------------------------------------------------------
// Selective scan: one wave per (b, k, channel-pair).  lane = {ci:1, n:4}.
// Fuses dt projection (rank 6) + softplus; butterfly-reduces h*C over n.
// ys[(b,k,c),l] = sum_n h*C + Ds*u   (Ds skip term folded in here)
// ---------------------------------------------------------------------------
__global__ void scan_kernel(const float* __restrict__ xd,
                            const float* __restrict__ xc_cm,
                            const float* __restrict__ dtw,
                            const float* __restrict__ dtb,
                            const float* __restrict__ Alog,
                            const float* __restrict__ Ds,
                            float* __restrict__ ys) {
  int gw   = blockIdx.x * 8 + (threadIdx.x >> 5);
  int lane = threadIdx.x & 31;
  int cpair = gw % (cfg::D / 2);
  int k     = (gw / (cfg::D / 2)) % 4;
  int b     = gw / ((cfg::D / 2) * 4);
  int n  = lane & 15;
  int ci = lane >> 4;
  int c  = cpair * 2 + ci;

  float Ac = -__expf(Alog[(size_t)(k * cfg::D + c) * cfg::N + n]);
  float w6[cfg::R];
#pragma unroll
  for (int r = 0; r < cfg::R; ++r) w6[r] = dtw[(size_t)(k * cfg::D + c) * cfg::R + r];
  float dtbv = dtb[k * cfg::D + c];
  float dsv  = Ds[k * cfg::D + c];

  const float* uptr = xc_cm + (size_t)(b * cfg::D + c) * cfg::L;
  const float* xrow = xd + (size_t)(b * 4 + k) * cfg::L * cfg::RN2;
  float* yout       = ys + (size_t)((b * 4 + k) * cfg::D + c) * cfg::L;

  float h = 0.f;
  for (int l = 0; l < cfg::L; ++l) {
    const float* row = xrow + (size_t)l * cfg::RN2;
    float dtr = dtbv;
#pragma unroll
    for (int r = 0; r < cfg::R; ++r) dtr += row[r] * w6[r];
    float dt = (dtr > 20.f) ? dtr : log1pf(__expf(dtr));  // softplus
    float u  = uptr[mapPos(k, l)];
    float Bn = row[cfg::R + n];
    float Cn = row[cfg::R + cfg::N + n];
    h = h * __expf(dt * Ac) + dt * u * Bn;
    float acc = h * Cn;
    acc += __shfl_xor(acc, 1, 32);
    acc += __shfl_xor(acc, 2, 32);
    acc += __shfl_xor(acc, 4, 32);
    acc += __shfl_xor(acc, 8, 32);
    if (n == 0) yout[l] = acc + dsv * u;
  }
}

// ---------------------------------------------------------------------------
// Combine 4 directions (with transpose/reversal gathers) + output LayerNorm
// + SiLU(z) gate -> bf16 rows for the out-proj GEMM.  Block = 192 threads.
// ---------------------------------------------------------------------------
__global__ void combine_ln_kernel(const float* __restrict__ ys,
                                  const float* __restrict__ z,
                                  const float* __restrict__ onw,
                                  const float* __restrict__ onb,
                                  bf16* __restrict__ tbuf) {
  int p = blockIdx.x;           // b*L + l
  int c = threadIdx.x;          // 0..191
  int b = p / cfg::L, l = p % cfg::L;
  int l1 = (l % cfg::W) * cfg::H + l / cfg::W;  // transposed-scan source index
  float v =
      ys[((size_t)((b * 4 + 0) * cfg::D + c)) * cfg::L + l] +
      ys[((size_t)((b * 4 + 1) * cfg::D + c)) * cfg::L + l1] +
      ys[((size_t)((b * 4 + 2) * cfg::D + c)) * cfg::L + (cfg::L - 1 - l)] +
      ys[((size_t)((b * 4 + 3) * cfg::D + c)) * cfg::L + (cfg::L - 1 - l1)];

  __shared__ float rs[6], rq[6], stat[2];
  float s = v, sq = v * v;
  for (int off = 16; off; off >>= 1) {
    s  += __shfl_xor(s,  off, 32);
    sq += __shfl_xor(sq, off, 32);
  }
  int wave = c >> 5, lane = c & 31;
  if (lane == 0) { rs[wave] = s; rq[wave] = sq; }
  __syncthreads();
  if (c == 0) {
    float ts = 0.f, tq = 0.f;
    for (int i = 0; i < 6; ++i) { ts += rs[i]; tq += rq[i]; }
    float m = ts / 192.f;
    stat[0] = m;
    stat[1] = rsqrtf(tq / 192.f - m * m + 1e-5f);
  }
  __syncthreads();
  float t  = (v - stat[0]) * stat[1] * onw[c] + onb[c];
  float zv = z[(size_t)p * cfg::D + c];
  t *= zv / (1.f + __expf(-zv));  // * silu(z)
  tbuf[(size_t)p * cfg::D + c] = (bf16)t;
}

// ---------------------------------------------------------------------------
// Host launcher
// ---------------------------------------------------------------------------
extern "C" void kernel_launch(void* const* d_in, const int* in_sizes, int n_in,
                              void* d_out, int out_size, void* d_ws, size_t ws_size,
                              hipStream_t stream) {
  using namespace cfg;
  (void)in_sizes; (void)n_in; (void)out_size; (void)ws_size;

  const float* x   = (const float*)d_in[0];
  float*       out = (float*)d_out;

  // workspace carve-up (all 256B aligned)
  auto align = [](size_t v) { return (v + 255) & ~size_t(255); };
  char* ws = (char*)d_ws;
  size_t off = 0;
  auto take = [&](size_t bytes) { char* p = ws + off; off += align(bytes); return p; };

  bf16*  xln   = (bf16*)take((size_t)M * DM * 2);          // LN'd input, bf16
  bf16*  wInB  = (bf16*)take((size_t)2 * D * DM * 2);      // in_w bf16
  bf16*  wXpB  = (bf16*)take((size_t)4 * RN2 * D * 2);     // xproj_w bf16
  bf16*  wOpB  = (bf16*)take((size_t)DM * D * 2);          // out_w bf16
  float* xzc   = (float*)take((size_t)M * D * 4);          // pre-conv xc (B,L,D)
  float* zbuf  = (float*)take((size_t)M * D * 4);          // z gate (B,L,D)
  bf16*  xc_rm = (bf16*)take((size_t)M * D * 2);           // conv out, bf16 (B,L,D)
  float* xc_cm = (float*)take((size_t)M * D * 4);          // conv out, f32 (B,D,L)
  float* xd    = (float*)take((size_t)B * 4 * L * RN2 * 4);// proj (B,4,L,38)
  float* ys    = (float*)take((size_t)B * 4 * D * L * 4);  // scan out (B,4,D,L)
  bf16*  tbuf  = (bf16*)take((size_t)M * D * 2);           // gated LN rows, bf16

  for (int br = 0; br < 2; ++br) {
    int base = 1 + br * 13;
    const float* lnw   = (const float*)d_in[base + 0];
    const float* lnb   = (const float*)d_in[base + 1];
    const float* inw   = (const float*)d_in[base + 2];
    const float* convw = (const float*)d_in[base + 3];
    const float* convb = (const float*)d_in[base + 4];
    const float* xpw   = (const float*)d_in[base + 5];
    const float* dtw   = (const float*)d_in[base + 6];
    const float* dtb   = (const float*)d_in[base + 7];
    const float* Alog  = (const float*)d_in[base + 8];
    const float* Dsp   = (const float*)d_in[base + 9];
    const float* onw   = (const float*)d_in[base + 10];
    const float* onb   = (const float*)d_in[base + 11];
    const float* opw   = (const float*)d_in[base + 12];
    const int brOff = br * DM;

    // 0) weights -> bf16
    {
      int n0 = 2 * D * DM, n1 = 4 * RN2 * D, n2 = DM * D;
      int nmax = n0 > n1 ? (n0 > n2 ? n0 : n2) : (n1 > n2 ? n1 : n2);
      cvt_bf16_kernel<<<(nmax + 255) / 256, 256, 0, stream>>>(
          inw, n0, xpw, n1, opw, n2, wInB, wXpB, wOpB);
    }
    // 1) LayerNorm -> bf16 rows
    ln_in_kernel<<<M / 8, 256, 0, stream>>>(x, lnw, lnb, xln, brOff);
    // 2) in-proj GEMM (9216 x 384 x 96) -> xc pre-conv + z   [TDM tiles]
    gemm_wmma_kernel<0><<<dim3(M / 128, (2 * D) / 64, 1), 256, 0, stream>>>(
        xln, DM, 2 * D, wInB, xzc, zbuf, nullptr, nullptr, 0);
    // 3) depthwise conv + SiLU (dual layout)
    conv_silu_kernel<<<(M * D) / 256, 256, 0, stream>>>(xzc, convw, convb, xc_rm, xc_cm);
    // 4) x-proj GEMM per direction (9216 x 38 x 192, A-rows remapped by k)
    gemm_wmma_kernel<1><<<dim3(M / 128, 1, 4), 256, 0, stream>>>(
        xc_rm, D, RN2, wXpB, xd, nullptr, nullptr, nullptr, 0);
    // 5) selective scan (1536 waves, 8 waves/block)
    scan_kernel<<<(B * 4 * (D / 2)) / 8, 256, 0, stream>>>(
        xd, xc_cm, dtw, dtb, Alog, Dsp, ys);
    // 6) combine 4 directions + LN + silu(z) gate -> bf16 rows
    combine_ln_kernel<<<M, 192, 0, stream>>>(ys, zbuf, onw, onb, tbuf);
    // 7) out-proj GEMM (9216 x 96 x 192) + residual into d_out half  [TDM tiles]
    gemm_wmma_kernel<2><<<dim3(M / 128, 2, 1), 256, 0, stream>>>(
        tbuf, D, DM, wOpB, nullptr, nullptr, x, out, brOff);
  }
}
